// JEPA_66434554135146
// MI455X (gfx1250) — matrix operand order repaired and checked by reading, not compiled, not run on hardware
//
#include <hip/hip_runtime.h>
#include <hip/hip_bf16.h>

typedef __attribute__((ext_vector_type(16))) _Float16 v16h;
typedef __attribute__((ext_vector_type(8)))  float    v8f;

#define DEVINL __device__ __forceinline__

// K-index pattern for the 16x32 f16 A fragment (matches B fills):
// lanes 0-15 hold row m=lane, elements 0..15 -> K=0..15;
// lanes 16-31 same rows with K=16..31.
DEVINL int a_kidx(int h, int hi) {
  return h + (hi << 4);
}

// A fragment from a row-major fp32 matrix (row = lane&15 of the 16-row tile).
DEVINL v16h load_a_rowmajor(const float* A, int ld, int row, int k0, int hi) {
  v16h a;
#pragma unroll
  for (int h = 0; h < 16; ++h)
    a[h] = (_Float16)A[(size_t)row * ld + k0 + a_kidx(h, hi)];
  return a;
}

// B fragment (32x16) representing W^T: B[k][n] = W[n][k]; lane (n=lane&15)
// reads 16 consecutive K from row n of row-major W -> fully contiguous.
DEVINL v16h load_b_wT(const float* W, int ld, int row, int k0, int hi) {
  const float* p = W + (size_t)row * ld + k0 + (hi << 4);
  v16h b;
#pragma unroll
  for (int h = 0; h < 16; ++h) b[h] = (_Float16)p[h];
  return b;
}

DEVINL v8f wmma_f16(v16h a, v16h b, v8f c) {
  return __builtin_amdgcn_wmma_f32_16x16x32_f16(false, a, false, b, (short)0, c,
                                                false, false);
}

DEVINL float sigm(float x) { return 1.0f / (1.0f + expf(-x)); }

// ---------------------------------------------------------------------------
// conv1 + BN + ReLU + 2x2 maxpool, direct VALU (K=18 too small for WMMA).
// in [512,2,65,65] -> out [512,32,32,32]
// ---------------------------------------------------------------------------
__global__ __launch_bounds__(256) void conv1_kernel(
    const float* __restrict__ x, const float* __restrict__ wgt,
    const float* __restrict__ cb, const float* __restrict__ g,
    const float* __restrict__ be, const float* __restrict__ mu,
    const float* __restrict__ var, float* __restrict__ y) {
  int gid = blockIdx.x * 256 + threadIdx.x;    // 512*32*32*32 threads
  int b = gid >> 15;
  int c = (gid >> 10) & 31;
  int pi = (gid >> 5) & 31;
  int pj = gid & 31;
  const float* xb = x + (size_t)b * 2 * 65 * 65;
  const float* wc = wgt + c * 18;
  float sc = g[c] * rsqrtf(var[c] + 1e-5f);
  float sh = (cb[c] - mu[c]) * sc + be[c];
  float best = 0.0f;  // relu floor
#pragma unroll
  for (int di = 0; di < 2; ++di)
#pragma unroll
    for (int dj = 0; dj < 2; ++dj) {
      int h = 2 * pi + di, w = 2 * pj + dj;
      float sum = 0.f;
#pragma unroll
      for (int ci = 0; ci < 2; ++ci)
#pragma unroll
        for (int r = 0; r < 3; ++r)
#pragma unroll
          for (int s = 0; s < 3; ++s) {
            int ih = h + r - 1, iw = w + s - 1;
            if ((unsigned)ih < 65u && (unsigned)iw < 65u)
              sum += xb[((size_t)ci * 65 + ih) * 65 + iw] * wc[ci * 9 + r * 3 + s];
          }
      best = fmaxf(best, fmaxf(sum * sc + sh, 0.f));
    }
  y[gid] = best;
}

// ---------------------------------------------------------------------------
// conv + BN + ReLU + 2x2 maxpool as WMMA implicit GEMM.
// Per wave: 16 output channels x 16 pre-pool pixels (2 rows x 8 cols strip),
// pooling done in lane space with shfl_xor(1)/shfl_xor(8).
// ---------------------------------------------------------------------------
template <int CIN, int HIN, int COUT>
__global__ __launch_bounds__(256) void conv_wmma_kernel(
    const float* __restrict__ x,   // [B,CIN,HIN,HIN]
    const float* __restrict__ wgt, // [COUT,CIN*9]
    const float* __restrict__ cb, const float* __restrict__ g,
    const float* __restrict__ be, const float* __restrict__ mu,
    const float* __restrict__ var,
    float* __restrict__ y) {       // [B,COUT,HIN/2,HIN/2]
  constexpr int K = CIN * 9;
  constexpr int NCHUNK = K / 32;
  constexpr int MT = COUT / 16;
  constexpr int TCOLS = HIN / 8;
  constexpr int PT = (HIN / 2) * TCOLS;
  const int lane = threadIdx.x & 31;
  const int wid = blockIdx.x * 8 + (threadIdx.x >> 5);
  const int tilesPerImg = MT * PT;
  const int b = wid / tilesPerImg;
  int rem = wid - b * tilesPerImg;
  const int mt = rem / PT;
  const int pt = rem - mt * PT;
  const int tr = pt / TCOLS;
  const int tc = pt - tr * TCOLS;
  const int ln = lane & 15, hi = lane >> 4;
  const int dr = ln >> 3, dc = ln & 7;
  const int ph = tr * 2 + dr, pw = tc * 8 + dc;
  const float* xb = x + (size_t)b * CIN * HIN * HIN;
  v8f acc = {};
  for (int kc = 0; kc < NCHUNK; ++kc) {
    const int k0 = kc * 32;
    // A = weight tile [16 x 32]
    v16h af;
#pragma unroll
    for (int h = 0; h < 16; ++h)
      af[h] = (_Float16)wgt[(size_t)(mt * 16 + ln) * K + k0 + a_kidx(h, hi)];
    // B = im2col tile [32 x 16]
    v16h bf;
#pragma unroll
    for (int h = 0; h < 16; ++h) {
      int kk = k0 + (hi << 4) + h;
      int ci = kk / 9;
      int r9 = kk - ci * 9;
      int r = r9 / 3;
      int s = r9 - r * 3;
      int ih = ph + r - 1, iw = pw + s - 1;
      float v = 0.f;
      if ((unsigned)ih < (unsigned)HIN && (unsigned)iw < (unsigned)HIN)
        v = xb[((size_t)ci * HIN + ih) * HIN + iw];
      bf[h] = (_Float16)v;
    }
    acc = wmma_f16(af, bf, acc);
  }
  // epilogue: BN + ReLU + 2x2 maxpool in lane space
  constexpr int HP = HIN / 2;
#pragma unroll
  for (int r = 0; r < 8; ++r) {
    int c = mt * 16 + r + (hi << 3);
    float sc = g[c] * rsqrtf(var[c] + 1e-5f);
    float v = (acc[r] + cb[c] - mu[c]) * sc + be[c];
    v = fmaxf(v, 0.f);
    v = fmaxf(v, __shfl_xor(v, 1, 32));  // pool cols
    v = fmaxf(v, __shfl_xor(v, 8, 32));  // pool rows
    if (dr == 0 && (dc & 1) == 0) {
      int pj = tc * 4 + (dc >> 1);
      y[(((size_t)b * COUT + c) * HP + tr) * HP + pj] = v;
    }
  }
}

// ---------------------------------------------------------------------------
// adaptive avg pool (8x8 -> 4x4): [512,128,8,8] -> flat [512,2048]
// ---------------------------------------------------------------------------
__global__ __launch_bounds__(256) void avgpool_kernel(const float* __restrict__ x3,
                                                      float* __restrict__ p) {
  int gid = blockIdx.x * 256 + threadIdx.x;  // 512*2048
  int b = gid >> 11;
  int k = gid & 2047;
  int c = k >> 4, yy = (k >> 2) & 3, xx = k & 3;
  const float* base = x3 + (((size_t)b * 128 + c) * 8 + yy * 2) * 8 + xx * 2;
  p[gid] = 0.25f * (base[0] + base[1] + base[8] + base[9]);
}

// ---------------------------------------------------------------------------
// s0 = p @ se_fc_w^T + b; writes s_cur and d_out[:,0,:]
// M=512,K=2048,N=256 -> 512 wave tiles, 64 blocks
// ---------------------------------------------------------------------------
__global__ __launch_bounds__(256) void fc_s0_kernel(
    const float* __restrict__ p, const float* __restrict__ w,
    const float* __restrict__ bias, float* __restrict__ s,
    float* __restrict__ out) {
  const int lane = threadIdx.x & 31;
  const int wid = blockIdx.x * 8 + (threadIdx.x >> 5);
  const int mt = wid >> 4;
  const int nt = wid & 15;
  const int ln = lane & 15, hi = lane >> 4;
  v8f acc = {};
  for (int kc = 0; kc < 64; ++kc) {
    const int k0 = kc * 32;
    v16h af = load_a_rowmajor(p, 2048, mt * 16 + ln, k0, hi);
    v16h bf = load_b_wT(w, 2048, nt * 16 + ln, k0, hi);
    acc = wmma_f16(af, bf, acc);
  }
#pragma unroll
  for (int r = 0; r < 8; ++r) {
    int row = mt * 16 + r + (hi << 3);
    int col = nt * 16 + ln;
    float v = acc[r] + bias[col];
    s[row * 256 + col] = v;
    out[(size_t)row * 4352 + col] = v;  // (T+1)*256 = 4352, slot t=0
  }
}

// ---------------------------------------------------------------------------
// Action encoder: relu(act @ w1^T + b1) @ w2^T + b2, fused.
// Block = 16 rows of [8192,*]; layer1 via VALU into LDS, layer2 via WMMA.
// ---------------------------------------------------------------------------
__global__ __launch_bounds__(256) void action_enc_kernel(
    const float* __restrict__ act, const float* __restrict__ w1,
    const float* __restrict__ b1, const float* __restrict__ w2,
    const float* __restrict__ b2, float* __restrict__ a) {
  __shared__ float a1[16 * 128];
  const int tid = threadIdx.x;
  const int rb = blockIdx.x * 16;
  for (int i = tid; i < 2048; i += 256) {
    int m = i >> 7, j = i & 127;
    int row = rb + m;
    float v = act[row * 2 + 0] * w1[j * 2 + 0] + act[row * 2 + 1] * w1[j * 2 + 1] + b1[j];
    a1[i] = fmaxf(v, 0.f);
  }
  __syncthreads();
  const int lane = tid & 31, wv = tid >> 5, ln = lane & 15, hi = lane >> 4;
  for (int q = 0; q < 2; ++q) {
    const int n0 = (wv * 2 + q) * 16;
    v8f acc = {};
    for (int kc = 0; kc < 4; ++kc) {
      const int k0 = kc * 32;
      v16h af = load_a_rowmajor(a1, 128, ln, k0, hi);
      v16h bf = load_b_wT(w2, 128, n0 + ln, k0, hi);
      acc = wmma_f16(af, bf, acc);
    }
#pragma unroll
    for (int r = 0; r < 8; ++r) {
      int row = rb + r + (hi << 3);
      int col = n0 + ln;
      a[(size_t)row * 256 + col] = acc[r] + b2[col];
    }
  }
}

// ---------------------------------------------------------------------------
// One rollout step, fully fused: x=[s||a_t] -> LSTM0 -> LSTM1 -> head -> out.
// Block = 16 batch rows (independent); chain flows through LDS.
// f gate skipped (c0 = 0 kills it).
// ---------------------------------------------------------------------------
__global__ __launch_bounds__(256) void rollout_step_kernel(
    const float* __restrict__ a, const float* __restrict__ l0w,
    const float* __restrict__ l0bi, const float* __restrict__ l0bh,
    const float* __restrict__ l1w, const float* __restrict__ l1bi,
    const float* __restrict__ l1bh, const float* __restrict__ pw1,
    const float* __restrict__ pb1, const float* __restrict__ pw2,
    const float* __restrict__ pb2, float* __restrict__ s,
    float* __restrict__ out, int t) {
  __shared__ float xb[16 * 512];
  __shared__ float h1[16 * 128];
  __shared__ float h2[16 * 128];
  __shared__ float f1[16 * 128];
  const int tid = threadIdx.x;
  const int m0 = blockIdx.x * 16;
  for (int i = tid; i < 8192; i += 256) {
    int m = i >> 9, k = i & 511;
    xb[i] = (k < 256) ? s[(size_t)(m0 + m) * 256 + k]
                      : a[((size_t)(m0 + m) * 16 + t) * 256 + (k - 256)];
  }
  __syncthreads();
  const int lane = tid & 31, wv = tid >> 5, ln = lane & 15, hi = lane >> 4;

  // ---- stage 1: gates0 (i,g,o) -> h1 ----
  {
    const int j0 = wv * 16;
    v8f ai = {}, ag = {}, ao = {};
    for (int kc = 0; kc < 16; ++kc) {
      const int k0 = kc * 32;
      v16h af = load_a_rowmajor(xb, 512, ln, k0, hi);
      v16h bi = load_b_wT(l0w, 512, j0 + ln, k0, hi);              // i rows 0..127
      v16h bg = load_b_wT(l0w + (size_t)256 * 512, 512, j0 + ln, k0, hi);  // g rows 256..
      v16h bo = load_b_wT(l0w + (size_t)384 * 512, 512, j0 + ln, k0, hi);  // o rows 384..
      ai = wmma_f16(af, bi, ai);
      ag = wmma_f16(af, bg, ag);
      ao = wmma_f16(af, bo, ao);
    }
#pragma unroll
    for (int r = 0; r < 8; ++r) {
      int mrow = r + (hi << 3);
      int col = j0 + ln;
      float gi = ai[r] + l0bi[col] + l0bh[col];
      float gg = ag[r] + l0bi[256 + col] + l0bh[256 + col];
      float go = ao[r] + l0bi[384 + col] + l0bh[384 + col];
      float cc = sigm(gi) * tanhf(gg);
      h1[mrow * 128 + col] = sigm(go) * tanhf(cc);
    }
  }
  __syncthreads();

  // ---- stage 2: gates1 (i,g,o) -> h2 ----
  {
    const int j0 = wv * 16;
    v8f ai = {}, ag = {}, ao = {};
    for (int kc = 0; kc < 4; ++kc) {
      const int k0 = kc * 32;
      v16h af = load_a_rowmajor(h1, 128, ln, k0, hi);
      v16h bi = load_b_wT(l1w, 128, j0 + ln, k0, hi);
      v16h bg = load_b_wT(l1w + (size_t)256 * 128, 128, j0 + ln, k0, hi);
      v16h bo = load_b_wT(l1w + (size_t)384 * 128, 128, j0 + ln, k0, hi);
      ai = wmma_f16(af, bi, ai);
      ag = wmma_f16(af, bg, ag);
      ao = wmma_f16(af, bo, ao);
    }
#pragma unroll
    for (int r = 0; r < 8; ++r) {
      int mrow = r + (hi << 3);
      int col = j0 + ln;
      float gi = ai[r] + l1bi[col] + l1bh[col];
      float gg = ag[r] + l1bi[256 + col] + l1bh[256 + col];
      float go = ao[r] + l1bi[384 + col] + l1bh[384 + col];
      float cc = sigm(gi) * tanhf(gg);
      h2[mrow * 128 + col] = sigm(go) * tanhf(cc);
    }
  }
  __syncthreads();

  // ---- stage 3: f1 = relu(h2 @ pw1^T + pb1) ----
  {
    const int j0 = wv * 16;
    v8f acc = {};
    for (int kc = 0; kc < 4; ++kc) {
      const int k0 = kc * 32;
      v16h af = load_a_rowmajor(h2, 128, ln, k0, hi);
      v16h bf = load_b_wT(pw1, 128, j0 + ln, k0, hi);
      acc = wmma_f16(af, bf, acc);
    }
#pragma unroll
    for (int r = 0; r < 8; ++r)
      f1[(r + (hi << 3)) * 128 + j0 + ln] = fmaxf(acc[r] + pb1[j0 + ln], 0.f);
  }
  __syncthreads();

  // ---- stage 4: out = f1 @ pw2^T + pb2 -> d_out slot t+1 and s (in place) ----
  for (int q = 0; q < 2; ++q) {
    const int n0 = (wv * 2 + q) * 16;
    v8f acc = {};
    for (int kc = 0; kc < 4; ++kc) {
      const int k0 = kc * 32;
      v16h af = load_a_rowmajor(f1, 128, ln, k0, hi);
      v16h bf = load_b_wT(pw2, 128, n0 + ln, k0, hi);
      acc = wmma_f16(af, bf, acc);
    }
#pragma unroll
    for (int r = 0; r < 8; ++r) {
      int row = m0 + r + (hi << 3);
      int col = n0 + ln;
      float v = acc[r] + pb2[col];
      out[(size_t)row * 4352 + (size_t)(t + 1) * 256 + col] = v;
      s[(size_t)row * 256 + col] = v;
    }
  }
}

// ---------------------------------------------------------------------------
extern "C" void kernel_launch(void* const* d_in, const int* in_sizes, int n_in,
                              void* d_out, int out_size, void* d_ws,
                              size_t ws_size, hipStream_t stream) {
  (void)in_sizes; (void)n_in; (void)out_size; (void)ws_size;
  const float* initial_state = (const float*)d_in[0];
  const float* actions = (const float*)d_in[1];
  const float* ae_w1 = (const float*)d_in[2];
  const float* ae_b1 = (const float*)d_in[3];
  const float* ae_w2 = (const float*)d_in[4];
  const float* ae_b2 = (const float*)d_in[5];
  const float* conv1_w = (const float*)d_in[6];
  const float* conv1_b = (const float*)d_in[7];
  const float* bn1_g = (const float*)d_in[8];
  const float* bn1_b = (const float*)d_in[9];
  const float* bn1_m = (const float*)d_in[10];
  const float* bn1_v = (const float*)d_in[11];
  const float* conv2_w = (const float*)d_in[12];
  const float* conv2_b = (const float*)d_in[13];
  const float* bn2_g = (const float*)d_in[14];
  const float* bn2_b = (const float*)d_in[15];
  const float* bn2_m = (const float*)d_in[16];
  const float* bn2_v = (const float*)d_in[17];
  const float* conv3_w = (const float*)d_in[18];
  const float* conv3_b = (const float*)d_in[19];
  const float* bn3_g = (const float*)d_in[20];
  const float* bn3_b = (const float*)d_in[21];
  const float* bn3_m = (const float*)d_in[22];
  const float* bn3_v = (const float*)d_in[23];
  const float* se_fc_w = (const float*)d_in[24];
  const float* se_fc_b = (const float*)d_in[25];
  const float* l0_wih = (const float*)d_in[26];
  const float* l0_bih = (const float*)d_in[28];
  const float* l0_bhh = (const float*)d_in[29];
  const float* l1_wih = (const float*)d_in[30];
  const float* l1_bih = (const float*)d_in[32];
  const float* l1_bhh = (const float*)d_in[33];
  const float* pf_w1 = (const float*)d_in[34];
  const float* pf_b1 = (const float*)d_in[35];
  const float* pf_w2 = (const float*)d_in[36];
  const float* pf_b2 = (const float*)d_in[37];

  float* out = (float*)d_out;
  float* ws = (float*)d_ws;
  // workspace layout (floats), with region reuse:
  float* x1 = ws;                         // [512,32,32,32] = 16777216
  float* x2 = ws + 16777216;              // [512,64,16,16] =  8388608
  float* x3 = ws;                         // reuse x1 region: [512,128,8,8] = 4194304
  float* p  = ws + 16777216;              // reuse x2 region: [512,2048]   = 1048576
  float* s  = ws + 25165824;              // [512,256]
  float* ab = ws + 25296896;              // [512*16,256]

  // 1. conv1+bn+relu+pool (VALU)
  conv1_kernel<<<65536, 256, 0, stream>>>(initial_state, conv1_w, conv1_b,
                                          bn1_g, bn1_b, bn1_m, bn1_v, x1);
  // 2. conv2 implicit-GEMM WMMA (512*4*64 wave tiles / 8 waves per block)
  conv_wmma_kernel<32, 32, 64><<<16384, 256, 0, stream>>>(
      x1, conv2_w, conv2_b, bn2_g, bn2_b, bn2_m, bn2_v, x2);
  // 3. conv3 implicit-GEMM WMMA (512*8*16 / 8)
  conv_wmma_kernel<64, 16, 128><<<8192, 256, 0, stream>>>(
      x2, conv3_w, conv3_b, bn3_g, bn3_b, bn3_m, bn3_v, x3);
  // 4. adaptive avg pool
  avgpool_kernel<<<4096, 256, 0, stream>>>(x3, p);
  // 5. FC -> s0 (also output slot 0)
  fc_s0_kernel<<<64, 256, 0, stream>>>(p, se_fc_w, se_fc_b, s, out);
  // 6. action encoder
  action_enc_kernel<<<512, 256, 0, stream>>>(actions, ae_w1, ae_b1, ae_w2,
                                             ae_b2, ab);
  // 7. sequential rollout, one fused kernel per step
  for (int t = 0; t < 16; ++t)
    rollout_step_kernel<<<32, 256, 0, stream>>>(
        ab, l0_wih, l0_bih, l0_bhh, l1_wih, l1_bih, l1_bhh, pf_w1, pf_b1,
        pf_w2, pf_b2, s, out, t);
}
